// TacticExpert_46136538693655
// MI455X (gfx1250) — matrix operand order repaired and checked by reading, not compiled, not run on hardware
//
#include <hip/hip_runtime.h>
#include <hip/hip_bf16.h>

#define E_     5
#define B_     32768
#define DIN_   768
#define DH_    384
#define DH2_   192
#define DOUT_  256
#define TAU_   0.07f
#define NW_    8      // waves per block (256 threads, wave32)
#define MT_    2      // 16-row M-tiles per block (32 tokens) -> halves weight/L2 traffic

typedef _Float16 h8  __attribute__((ext_vector_type(8)));
typedef _Float16 h16 __attribute__((ext_vector_type(16)));
typedef float    v8f __attribute__((ext_vector_type(8)));

static __device__ inline h16 cmb(h8 a, h8 b) {
    return __builtin_shufflevector(a, b, 0,1,2,3,4,5,6,7,8,9,10,11,12,13,14,15);
}
static __device__ inline v8f vzero() {
    v8f z;
#pragma unroll
    for (int i = 0; i < 8; ++i) z[i] = 0.0f;
    return z;
}
// Load one A-fragment (16x32 f16 layout) from a global f32 row, converting in regs.
static __device__ inline h16 loadA_f32(const float* __restrict__ xr, int k0, int hi) {
    const float4* p0 = (const float4*)(xr + k0 + hi * 8);
    const float4* p1 = (const float4*)(xr + k0 + 16 + hi * 8);
    float4 f0 = p0[0], f1 = p0[1], f2 = p1[0], f3 = p1[1];
    h16 a;
    a[0]=(_Float16)f0.x; a[1]=(_Float16)f0.y; a[2]=(_Float16)f0.z; a[3]=(_Float16)f0.w;
    a[4]=(_Float16)f1.x; a[5]=(_Float16)f1.y; a[6]=(_Float16)f1.z; a[7]=(_Float16)f1.w;
    a[8]=(_Float16)f2.x; a[9]=(_Float16)f2.y; a[10]=(_Float16)f2.z; a[11]=(_Float16)f2.w;
    a[12]=(_Float16)f3.x; a[13]=(_Float16)f3.y; a[14]=(_Float16)f3.z; a[15]=(_Float16)f3.w;
    return a;
}

// ---------------- weight conversion ----------------
__global__ void cvt_f32_f16(const float* __restrict__ src, _Float16* __restrict__ dst, int n) {
    int i = blockIdx.x * blockDim.x + threadIdx.x;
    int s = gridDim.x * blockDim.x;
    for (; i < n; i += s) dst[i] = (_Float16)src[i];
}
// extract V slice (rows 512..767) of each expert's wqkv [E][768][256] -> [E][256][256]
__global__ void cvt_wv(const float* __restrict__ wqkv, _Float16* __restrict__ wv, int n) {
    int i = blockIdx.x * blockDim.x + threadIdx.x;
    int s = gridDim.x * blockDim.x;
    for (; i < n; i += s) {
        int e = i >> 16;
        int rc = i & 65535;
        int r = rc >> 8, c = rc & 255;
        wv[i] = (_Float16)wqkv[(size_t)e * (768 * 256) + (size_t)(512 + r) * 256 + c];
    }
}
__global__ void zero_counts(int* c) { if (threadIdx.x < 8) c[threadIdx.x] = 0; }

// ---------------- GEMM helpers (2 M-tiles share every B-fragment) ----------------
// Layer-1: A from global f32 x rows (cvt in regs), K-outer so x is read exactly once.
// Covers N = NW_*NT*16 outputs; W f16 [N][768] row-major; raw f32 -> sRaw (stride rs).
template<int NT>
static __device__ inline void gemm_x2(const float* __restrict__ x, const int* __restrict__ tok,
                                      const _Float16* __restrict__ W,
                                      float* sRaw, int rs, int wave, int lane) {
    v8f acc[NT][MT_];
#pragma unroll
    for (int t = 0; t < NT; ++t)
#pragma unroll
        for (int mt = 0; mt < MT_; ++mt) acc[t][mt] = vzero();
    const int m = lane & 15, hi = lane >> 4;
    const float* xr0 = x + (size_t)tok[m] * DIN_;
    const float* xr1 = x + (size_t)tok[16 + m] * DIN_;
    for (int k0 = 0; k0 < DIN_; k0 += 32) {
        h16 a0 = loadA_f32(xr0, k0, hi);
        h16 a1 = loadA_f32(xr1, k0, hi);
#pragma unroll
        for (int t = 0; t < NT; ++t) {
            int n0 = (wave * NT + t) * 16;
            const _Float16* wr = W + (size_t)(n0 + m) * DIN_ + k0 + hi * 8;
            h16 b = cmb(*(const h8*)wr, *(const h8*)(wr + 16));
            acc[t][0] = __builtin_amdgcn_wmma_f32_16x16x32_f16(false, a0, false, b,
                                                               (short)0, acc[t][0], false, false);
            acc[t][1] = __builtin_amdgcn_wmma_f32_16x16x32_f16(false, a1, false, b,
                                                               (short)0, acc[t][1], false, false);
        }
    }
#pragma unroll
    for (int t = 0; t < NT; ++t) {
        int n0 = (wave * NT + t) * 16;
#pragma unroll
        for (int mt = 0; mt < MT_; ++mt)
#pragma unroll
            for (int r = 0; r < 8; ++r)
                sRaw[(mt * 16 + hi * 8 + r) * rs + n0 + m] = acc[t][mt][r];
    }
}

// Generic: A f16 in LDS (32 x astride), W f16 [N][ldw] global, raw f32 out (32 rows).
static __device__ inline void gemm_lds2(const _Float16* __restrict__ A, int astride,
                                        const _Float16* __restrict__ W, int ldw,
                                        int K, int N, float* sRaw, int rs, int wave, int lane) {
    const int n = lane & 15, hi = lane >> 4;
    for (int t = wave; t < (N >> 4); t += NW_) {
        int n0 = t << 4;
        v8f acc0 = vzero(), acc1 = vzero();
        const _Float16* wr = W + (size_t)(n0 + n) * ldw;
        for (int k0 = 0; k0 < K; k0 += 32) {
            h16 b = cmb(*(const h8*)(wr + k0 + hi * 8),
                        *(const h8*)(wr + k0 + 16 + hi * 8));
            h16 a0 = cmb(*(const h8*)(A + n * astride + k0 + hi * 8),
                         *(const h8*)(A + n * astride + k0 + 16 + hi * 8));
            h16 a1 = cmb(*(const h8*)(A + (16 + n) * astride + k0 + hi * 8),
                         *(const h8*)(A + (16 + n) * astride + k0 + 16 + hi * 8));
            acc0 = __builtin_amdgcn_wmma_f32_16x16x32_f16(false, a0, false, b,
                                                          (short)0, acc0, false, false);
            acc1 = __builtin_amdgcn_wmma_f32_16x16x32_f16(false, a1, false, b,
                                                          (short)0, acc1, false, false);
        }
#pragma unroll
        for (int r = 0; r < 8; ++r) {
            sRaw[(hi * 8 + r) * rs + n0 + n]        = acc0[r];
            sRaw[(16 + hi * 8 + r) * rs + n0 + n]   = acc1[r];
        }
    }
}

// LayerNorm(+opt ReLU) over width W per row; 32 rows, 8 threads/row, shuffle reduction.
static __device__ inline void ln_act(float* sRaw, int rs, const float* __restrict__ bias,
                                     const float* __restrict__ g, const float* __restrict__ be,
                                     int W, bool relu,
                                     _Float16* outH, int hs, float* outF, int fs, int tid) {
    int row = tid >> 3, l8 = tid & 7;
    float s1 = 0.f, s2 = 0.f;
    for (int c = l8; c < W; c += 8) {
        float v = sRaw[row * rs + c] + bias[c];
        s1 += v; s2 += v * v;
    }
    for (int off = 1; off < 8; off <<= 1) {
        s1 += __shfl_xor(s1, off, 32);
        s2 += __shfl_xor(s2, off, 32);
    }
    float mu = s1 / (float)W;
    float var = s2 / (float)W - mu * mu;
    float rinv = rsqrtf(var + 1e-5f);
    for (int c = l8; c < W; c += 8) {
        float v = sRaw[row * rs + c] + bias[c];
        float o = (v - mu) * rinv * g[c] + be[c];
        if (relu) o = fmaxf(o, 0.f);
        if (outH) outH[row * hs + c] = (_Float16)o;
        if (outF) outF[row * fs + c] = o;
    }
}

// ---------------- router (32 tokens per block, 256 threads) ----------------
__global__ __launch_bounds__(256) void router_kernel(
    const float* __restrict__ x, const float* __restrict__ gumbel,
    const _Float16* __restrict__ rw1f, const float* __restrict__ rb1,
    const float* __restrict__ rg1, const float* __restrict__ rbe1,
    const _Float16* __restrict__ rw2f, const float* __restrict__ rb2,
    const float* __restrict__ rg2, const float* __restrict__ rbe2,
    const float* __restrict__ rw3, const float* __restrict__ rb3,
    float* __restrict__ out_rw, float* __restrict__ out_idx,
    int* __restrict__ counts, int* __restrict__ lists)
{
    __shared__ __align__(16) float    sRaw[32 * 392];
    __shared__ __align__(16) _Float16 sHa[32 * 392];
    __shared__ __align__(16) float    sLog[32 * 5];
    __shared__ int sTok[32];

    const int tid = threadIdx.x, lane = tid & 31, wave = tid >> 5;
    if (tid < 32) sTok[tid] = blockIdx.x * 32 + tid;
    __syncthreads();

    gemm_x2<3>(x, sTok, rw1f, sRaw, 392, wave, lane);                 // [32,384]
    __syncthreads();
    ln_act(sRaw, 392, rb1, rg1, rbe1, DH_, true, sHa, 392, nullptr, 0, tid);
    __syncthreads();
    gemm_lds2(sHa, 392, rw2f, DH_, DH_, DH2_, sRaw, 392, wave, lane); // [32,192]
    __syncthreads();
    ln_act(sRaw, 392, rb2, rg2, rbe2, DH2_, true, nullptr, 0, sRaw, 392, tid); // f32 in place
    __syncthreads();

    for (int i = tid; i < 32 * E_; i += 256) {                        // logits [32,5]
        int t = i / E_, e = i - t * E_;
        float s = rb3[e];
        const float* wr = rw3 + (size_t)e * DH2_;
        for (int c = 0; c < DH2_; ++c) s += sRaw[t * 392 + c] * wr[c];
        sLog[t * E_ + e] = s;
    }
    __syncthreads();

    if (tid < 32) {
        int tok = blockIdx.x * 32 + tid;
        float z[E_];
        float zmax = -3.4e38f; int arg = 0;
        for (int e = 0; e < E_; ++e) {
            z[e] = (sLog[tid * E_ + e] + gumbel[(size_t)tok * E_ + e]) / TAU_;
            if (z[e] > zmax) { zmax = z[e]; arg = e; }
        }
        float se = 0.f;
        for (int e = 0; e < E_; ++e) { z[e] = expf(z[e] - zmax); se += z[e]; }
        for (int e = 0; e < E_; ++e) {
            float s = z[e] / se;
            float hard = (e == arg) ? 1.f : 0.f;
            out_rw[(size_t)tok * E_ + e] = (hard + s) - s;            // straight-through fwd
        }
        out_idx[tok] = (float)arg;
        int slot = atomicAdd(&counts[arg], 1);
        lists[arg * B_ + slot] = tok;
    }
}

// ---------------- experts (routed, 32-token tiles, 256 threads) ----------------
__global__ __launch_bounds__(256) void expert_kernel(
    const float* __restrict__ x, const int* __restrict__ counts, const int* __restrict__ lists,
    const _Float16* __restrict__ ew1f, const float* __restrict__ eb1,
    const float* __restrict__ eg1, const float* __restrict__ ebe1,
    const _Float16* __restrict__ ew2f, const float* __restrict__ eb2,
    const float* __restrict__ eg2, const float* __restrict__ ebe2,
    const _Float16* __restrict__ ew3f, const float* __restrict__ eb3,
    const float* __restrict__ eg3, const float* __restrict__ ebe3,
    const _Float16* __restrict__ wvf, const float* __restrict__ bqkv,
    const _Float16* __restrict__ wof, const float* __restrict__ bo,
    const float* __restrict__ ng, const float* __restrict__ nb,
    float* __restrict__ out)
{
    __shared__ __align__(16) float    sRaw[32 * 392];
    __shared__ __align__(16) _Float16 sHa[32 * 392];
    __shared__ __align__(16) _Float16 sHb[32 * 392];
    __shared__ __align__(16) float    sY[32 * 256];
    __shared__ int sTok[32];

    const int tid = threadIdx.x, lane = tid & 31, wave = tid >> 5;

    // map block -> (expert e, tile of 32 tokens)
    int b = blockIdx.x, e = 0, acc = 0;
    for (e = 0; e < E_; ++e) {
        int te = (counts[e] + 31) >> 5;
        if (b < acc + te) break;
        acc += te;
    }
    if (e >= E_) return;
    int tile = b - acc;
    int cnt = counts[e];
    int nvalid = cnt - tile * 32; if (nvalid > 32) nvalid = 32;
    if (tid < 32)
        sTok[tid] = (tid < nvalid) ? lists[e * B_ + tile * 32 + tid]
                                   : lists[e * B_ + tile * 32];      // pad w/ dup valid token
    __syncthreads();

    const size_t eo384 = (size_t)e * 384, eo256 = (size_t)e * 256;

    gemm_x2<3>(x, sTok, ew1f + (size_t)e * DH_ * DIN_, sRaw, 392, wave, lane);     // 768->384
    __syncthreads();
    ln_act(sRaw, 392, eb1 + eo384, eg1 + eo384, ebe1 + eo384, DH_, true, sHa, 392, nullptr, 0, tid);
    __syncthreads();
    gemm_lds2(sHa, 392, ew2f + (size_t)e * DH_ * DH_, DH_, DH_, DH_, sRaw, 392, wave, lane); // 384->384
    __syncthreads();
    ln_act(sRaw, 392, eb2 + eo384, eg2 + eo384, ebe2 + eo384, DH_, true, sHb, 392, nullptr, 0, tid);
    __syncthreads();
    gemm_lds2(sHb, 392, ew3f + (size_t)e * DOUT_ * DH_, DH_, DH_, DOUT_, sRaw, 392, wave, lane); // 384->256
    __syncthreads();
    ln_act(sRaw, 392, eb3 + eo256, eg3 + eo256, ebe3 + eo256, DOUT_, false, sHa, 392, sY, 256, tid); // y
    __syncthreads();
    gemm_lds2(sHa, 392, wvf + (size_t)e * DOUT_ * DOUT_, DOUT_, DOUT_, DOUT_, sRaw, 392, wave, lane); // v=y@Wv^T
    __syncthreads();
    {   // v += bv ; to f16  (softmax over length-1 seq == 1, so attn == v; q,k dropped)
        int row = tid >> 3, l8 = tid & 7;
        const float* bv = bqkv + (size_t)e * 768 + 512;
        for (int c = l8; c < DOUT_; c += 8)
            sHb[row * 392 + c] = (_Float16)(sRaw[row * 392 + c] + bv[c]);
    }
    __syncthreads();
    gemm_lds2(sHb, 392, wof + (size_t)e * DOUT_ * DOUT_, DOUT_, DOUT_, DOUT_, sRaw, 392, wave, lane); // out=v@Wo^T
    __syncthreads();
    {   // final LN(y + out + bo) -> d_out
        int row = tid >> 3, l8 = tid & 7;
        const float* bop = bo + eo256;
        const float* gp = ng + eo256;
        const float* bp = nb + eo256;
        float s1 = 0.f, s2 = 0.f;
        for (int c = l8; c < DOUT_; c += 8) {
            float v = sY[row * 256 + c] + sRaw[row * 392 + c] + bop[c];
            s1 += v; s2 += v * v;
        }
        for (int off = 1; off < 8; off <<= 1) {
            s1 += __shfl_xor(s1, off, 32);
            s2 += __shfl_xor(s2, off, 32);
        }
        float mu = s1 / (float)DOUT_;
        float var = s2 / (float)DOUT_ - mu * mu;
        float rinv = rsqrtf(var + 1e-5f);
        if (row < nvalid) {
            float* orow = out + (size_t)sTok[row] * DOUT_;
            for (int c = l8; c < DOUT_; c += 8) {
                float v = sY[row * 256 + c] + sRaw[row * 392 + c] + bop[c];
                orow[c] = (v - mu) * rinv * gp[c] + bp[c];
            }
        }
    }
}

// ---------------- launcher ----------------
extern "C" void kernel_launch(void* const* d_in, const int* in_sizes, int n_in,
                              void* d_out, int out_size, void* d_ws, size_t ws_size,
                              hipStream_t stream) {
    (void)in_sizes; (void)n_in; (void)out_size; (void)ws_size;
    const float* x     = (const float*)d_in[0];
    const float* gum   = (const float*)d_in[1];
    const float* rw1   = (const float*)d_in[2];
    const float* rb1   = (const float*)d_in[3];
    const float* rg1   = (const float*)d_in[4];
    const float* rbe1  = (const float*)d_in[5];
    const float* rw2   = (const float*)d_in[6];
    const float* rb2   = (const float*)d_in[7];
    const float* rg2   = (const float*)d_in[8];
    const float* rbe2  = (const float*)d_in[9];
    const float* rw3   = (const float*)d_in[10];
    const float* rb3   = (const float*)d_in[11];
    const float* ew1   = (const float*)d_in[12];
    const float* eb1   = (const float*)d_in[13];
    const float* eg1   = (const float*)d_in[14];
    const float* ebe1  = (const float*)d_in[15];
    const float* ew2   = (const float*)d_in[16];
    const float* eb2   = (const float*)d_in[17];
    const float* eg2   = (const float*)d_in[18];
    const float* ebe2  = (const float*)d_in[19];
    const float* ew3   = (const float*)d_in[20];
    const float* eb3   = (const float*)d_in[21];
    const float* eg3   = (const float*)d_in[22];
    const float* ebe3  = (const float*)d_in[23];
    const float* wqkv  = (const float*)d_in[24];
    const float* bqkv  = (const float*)d_in[25];
    const float* wo    = (const float*)d_in[26];
    const float* bo    = (const float*)d_in[27];
    const float* ng    = (const float*)d_in[28];
    const float* nb    = (const float*)d_in[29];

    float* out_y   = (float*)d_out;                        // [B,256]
    float* out_rw  = out_y + (size_t)B_ * DOUT_;           // [B,5]
    float* out_idx = out_rw + (size_t)B_ * E_;             // [B]

    char* wsb = (char*)d_ws;
    size_t off = 0;
    auto carve = [&](size_t bytes) -> char* {
        char* p = wsb + off;
        off = (off + bytes + 255) & ~(size_t)255;
        return p;
    };
    int* counts = (int*)carve(16 * sizeof(int));
    int* lists  = (int*)carve((size_t)E_ * B_ * sizeof(int));
    _Float16* rw1f = (_Float16*)carve((size_t)DH_ * DIN_ * 2);
    _Float16* rw2f = (_Float16*)carve((size_t)DH2_ * DH_ * 2);
    _Float16* ew1f = (_Float16*)carve((size_t)E_ * DH_ * DIN_ * 2);
    _Float16* ew2f = (_Float16*)carve((size_t)E_ * DH_ * DH_ * 2);
    _Float16* ew3f = (_Float16*)carve((size_t)E_ * DOUT_ * DH_ * 2);
    _Float16* wvf  = (_Float16*)carve((size_t)E_ * DOUT_ * DOUT_ * 2);
    _Float16* wof  = (_Float16*)carve((size_t)E_ * DOUT_ * DOUT_ * 2);

    // weight conversions (cheap: ~4M elements total)
    cvt_f32_f16<<<1024, 256, 0, stream>>>(rw1, rw1f, DH_ * DIN_);
    cvt_f32_f16<<<256,  256, 0, stream>>>(rw2, rw2f, DH2_ * DH_);
    cvt_f32_f16<<<2048, 256, 0, stream>>>(ew1, ew1f, E_ * DH_ * DIN_);
    cvt_f32_f16<<<1024, 256, 0, stream>>>(ew2, ew2f, E_ * DH_ * DH_);
    cvt_f32_f16<<<1024, 256, 0, stream>>>(ew3, ew3f, E_ * DOUT_ * DH_);
    cvt_wv    <<<512,  256, 0, stream>>>(wqkv, wvf, E_ * DOUT_ * DOUT_);
    cvt_f32_f16<<<512,  256, 0, stream>>>(wo, wof, E_ * DOUT_ * DOUT_);
    zero_counts<<<1, 32, 0, stream>>>(counts);

    router_kernel<<<B_ / 32, 256, 0, stream>>>(
        x, gum, rw1f, rb1, rg1, rbe1, rw2f, rb2, rg2, rbe2, rw3, rb3,
        out_rw, out_idx, counts, lists);

    // worst case tiles = 1024 + (E-1) partials; 1029 is a safe upper bound
    expert_kernel<<<1029, 256, 0, stream>>>(
        x, counts, lists,
        ew1f, eb1, eg1, ebe1, ew2f, eb2, eg2, ebe2, ew3f, eb3, eg3, ebe3,
        wvf, bqkv, wof, bo, ng, nb, out_y);
}
